// WarpBlock_47141561040963
// MI455X (gfx1250) — compile-verified, matrix operand order checked by reading, not run on hardware
//
#include <hip/hip_runtime.h>
#include <hip/hip_bf16.h>

// ---------------------------------------------------------------------------
// Modulated deformable conv pipeline for MI455X (gfx1250, wave32, WMMA).
// Three implicit GEMMs, M=36864, K=1152, N=128/108/128 using
// v_wmma_f32_16x16x32_f16 (f16 operands, f32 accumulate).
// Block tile: 32 pixels x 128 out ch, 8 waves, double-buffered LDS A tile,
// one barrier per K chunk, 2 WMMAs per wave per chunk (shared B fragment).
// ---------------------------------------------------------------------------

typedef __attribute__((ext_vector_type(16))) _Float16 v16h;
typedef __attribute__((ext_vector_type(8)))  _Float16 v8h;
typedef __attribute__((ext_vector_type(8)))  float    v8f;

#define CH   128
#define Hh   96
#define Ww   96
#define Bn   4
#define HP   98           // padded spatial (96 + 1 halo each side)
#define WP   98
#define KCH  36           // K chunks of 32 -> K = 1152
#define NOFF 108          // conv2 real output channels
#define MT   32           // pixels per block tile

union H4 { uint2 u; _Float16 h[4]; };

// ---------------------------------------------------------------------------
// Pack x (NCHW f32) -> padded NHWC f16 with zero halo.
// ---------------------------------------------------------------------------
__global__ __launch_bounds__(256) void pack_x_kernel(const float* __restrict__ x,
                                                     _Float16* __restrict__ xh) {
    int idx = blockIdx.x * 256 + threadIdx.x;               // < 4*98*98*128
    int c   = idx & (CH - 1);
    int r   = idx >> 7;
    int wp  = r % WP;
    int r2  = r / WP;
    int hp  = r2 % HP;
    int b   = r2 / HP;
    if (b >= Bn) return;
    float v = 0.0f;
    if (hp >= 1 && hp <= Hh && wp >= 1 && wp <= Ww)
        v = x[(((size_t)(b * CH + c) * Hh) + (hp - 1)) * Ww + (wp - 1)];
    xh[idx] = (_Float16)v;
}

// ---------------------------------------------------------------------------
// Pack weights (O,128,3,3) f32 into WMMA B-fragment order:
//   wpack[kc][nc][lane][j], 16 halves per lane (one v16h load).
// Fragment K mapping (16-bit A/B layout):
//   kbase = (lane>=16)?8:0 ;  K_local = (j<8 ? 0 : 16) + kbase + (j&7)
// mode 0 (conv): kc -> tap = kc>>2, ci = (kc&3)*32 + K_local
// mode 1 (dcn) : kc -> g = kc/9, k = kc%9, ci = g*32 + K_local
// ---------------------------------------------------------------------------
__global__ __launch_bounds__(256) void pack_w_kernel(const float* __restrict__ w,
                                                     _Float16* __restrict__ wpack,
                                                     int mode, int Osrc) {
    int tid  = blockIdx.x * 256 + threadIdx.x;              // < 36*8*32 = 9216
    int l    = tid & 31;
    int nc   = (tid >> 5) & 7;
    int kc   = tid >> 8;
    int n    = nc * 16 + (l & 15);
    int kb   = (l >> 4) << 3;
#pragma unroll
    for (int j = 0; j < 16; ++j) {
        int kl = ((j < 8) ? 0 : 16) + kb + (j & 7);
        int ci, tap;
        if (mode == 0) { tap = kc >> 2; ci = ((kc & 3) << 5) + kl; }
        else           { int g = kc / 9; tap = kc - g * 9; ci = (g << 5) + kl; }
        float v = (n < Osrc) ? w[(size_t)n * 1152 + ci * 9 + tap] : 0.0f;
        wpack[(size_t)tid * 16 + j] = (_Float16)v;
    }
}

// ---------------------------------------------------------------------------
// 3x3 conv as implicit GEMM, 32-pixel x 128-ch tile, double-buffered A.
// mode 0: out -> padded NHWC f16.  mode 1: out -> [pix][108] f32.
// ---------------------------------------------------------------------------
__global__ __launch_bounds__(256) void conv_gemm_kernel(const _Float16* __restrict__ Ain,
                                                        const _Float16* __restrict__ wpack,
                                                        const float* __restrict__ bias,
                                                        _Float16* __restrict__ out16,
                                                        float* __restrict__ outco,
                                                        int mode, int Ob) {
    __shared__ _Float16 Alds[2][MT * 32];
    int tid  = threadIdx.x;
    int wave = tid >> 5, lane = tid & 31;
    int blk  = blockIdx.x;                                  // 4*96*3
    int wt   = blk % 3;
    int h    = (blk / 3) % Hh;
    int b    = blk / (3 * Hh);
    int w0   = wt * MT;

    int row   = lane & 15;
    int kbase = (lane >> 4) << 3;
    int n     = wave * 16 + row;

    v8f c0, c1;
    float bv = (n < Ob) ? bias[n] : 0.0f;
#pragma unroll
    for (int v = 0; v < 8; ++v) { c0[v] = bv; c1[v] = bv; }

    int p  = tid >> 3;                                      // fill pixel 0..31
    int c4 = (tid & 7) << 2;                                // fill channel quad

    auto loadA = [&](int kc) -> uint2 {
        int tap = kc >> 2;
        int dh  = tap / 3, dw = tap - dh * 3;
        int ci  = ((kc & 3) << 5) + c4;
        const _Float16* src = Ain +
            (((size_t)(b * HP + h + dh) * WP) + (w0 + p + dw)) * CH + ci;
        return *(const uint2*)src;                          // 4 halves
    };

    uint2 v0 = loadA(0);
    *(uint2*)&Alds[0][(p << 5) + c4] = v0;
    __syncthreads();

    for (int kc = 0; kc < KCH; ++kc) {
        int cur = kc & 1;
        uint2 nxt;
        if (kc < KCH - 1) nxt = loadA(kc + 1);

        v8h s0 = *(const v8h*)&Alds[cur][(row << 5) + kbase];
        v8h s1 = *(const v8h*)&Alds[cur][(row << 5) + 16 + kbase];
        v8h t0 = *(const v8h*)&Alds[cur][((row + 16) << 5) + kbase];
        v8h t1 = *(const v8h*)&Alds[cur][((row + 16) << 5) + 16 + kbase];
        v16h a0, a1;
#pragma unroll
        for (int j = 0; j < 8; ++j) {
            a0[j] = s0[j]; a0[j + 8] = s1[j];
            a1[j] = t0[j]; a1[j + 8] = t1[j];
        }
        v16h bf = *(const v16h*)(wpack + ((((size_t)kc << 3) + wave) * 32 + lane) * 16);
        c0 = __builtin_amdgcn_wmma_f32_16x16x32_f16(false, a0, false, bf,
                                                    (short)0, c0, false, false);
        c1 = __builtin_amdgcn_wmma_f32_16x16x32_f16(false, a1, false, bf,
                                                    (short)0, c1, false, false);
        if (kc < KCH - 1) *(uint2*)&Alds[cur ^ 1][(p << 5) + c4] = nxt;
        __syncthreads();
    }

    int mlo = (lane >> 4) << 3;
    if (mode == 0) {
#pragma unroll
        for (int v = 0; v < 8; ++v) {
            int m = v + mlo;
            size_t base = (((size_t)(b * HP + h + 1) * WP) + (w0 + m + 1)) * CH + n;
            out16[base] = (_Float16)c0[v];
            out16[base + 16 * CH] = (_Float16)c1[v];        // pixel m+16
        }
    } else if (n < NOFF) {
#pragma unroll
        for (int v = 0; v < 8; ++v) {
            int m = v + mlo;
            size_t base = (((size_t)(b * Hh + h) * Ww) + (w0 + m)) * NOFF + n;
            outco[base] = c0[v];
            outco[base + 16 * NOFF] = c1[v];                // pixel m+16
        }
    }
}

// ---------------------------------------------------------------------------
// Modulated deformable conv: per K chunk (g,k) bilinear-gather 32 channels,
// modulate by sigmoid(mask), double-buffered WMMA loop, NCHW f32 output via
// LDS transpose.
// ---------------------------------------------------------------------------
__global__ __launch_bounds__(256) void dcn_gemm_kernel(const _Float16* __restrict__ xh,
                                                       const float* __restrict__ co,
                                                       const _Float16* __restrict__ wpack,
                                                       const float* __restrict__ bias,
                                                       float* __restrict__ out) {
    __shared__ _Float16 Alds[2][MT * 32];
    __shared__ float    Olds[MT * CH];
    int tid  = threadIdx.x;
    int wave = tid >> 5, lane = tid & 31;
    int blk  = blockIdx.x;
    int wt   = blk % 3;
    int h    = (blk / 3) % Hh;
    int b    = blk / (3 * Hh);
    int w0   = wt * MT;

    int row   = lane & 15;
    int kbase = (lane >> 4) << 3;
    int n     = wave * 16 + row;

    v8f c0, c1;
    float bv = bias[n];
#pragma unroll
    for (int v = 0; v < 8; ++v) { c0[v] = bv; c1[v] = bv; }

    int p  = tid >> 3;
    int c4 = (tid & 7) << 2;
    int wq = w0 + p;
    const float* cop = co + (((size_t)(b * Hh + h) * Ww) + wq) * NOFF;

    auto gatherA = [&](int kc) -> uint2 {
        int g = kc / 9, k = kc - g * 9;
        float offy = cop[g * 18 + k * 2 + 0];
        float offx = cop[g * 18 + k * 2 + 1];
        float msk  = 1.0f / (1.0f + __expf(-cop[72 + g * 9 + k]));
        float ys = (float)h  + (float)(k / 3 - 1) + offy;
        float xs = (float)wq + (float)(k % 3 - 1) + offx;
        float y0 = floorf(ys), x0 = floorf(xs);
        float fy = ys - y0,    fx = xs - x0;
        int y0i = (int)y0, x0i = (int)x0;
        float wy[2] = {1.0f - fy, fy};
        float wx[2] = {1.0f - fx, fx};
        float acc[4] = {0.0f, 0.0f, 0.0f, 0.0f};
#pragma unroll
        for (int dy = 0; dy < 2; ++dy) {
#pragma unroll
            for (int dx = 0; dx < 2; ++dx) {
                int yy = y0i + dy, xx = x0i + dx;
                bool valid = (yy >= 0) & (yy < Hh) & (xx >= 0) & (xx < Ww);
                int yc = yy < 0 ? 0 : (yy > Hh - 1 ? Hh - 1 : yy);
                int xc = xx < 0 ? 0 : (xx > Ww - 1 ? Ww - 1 : xx);
                H4 v;
                v.u = *(const uint2*)(xh +
                    (((size_t)(b * HP + yc + 1) * WP) + (xc + 1)) * CH +
                    (g << 5) + c4);
                float wgt = valid ? wy[dy] * wx[dx] : 0.0f;
#pragma unroll
                for (int q = 0; q < 4; ++q) acc[q] += wgt * (float)v.h[q];
            }
        }
        H4 r;
#pragma unroll
        for (int q = 0; q < 4; ++q) r.h[q] = (_Float16)(acc[q] * msk);
        return r.u;
    };

    uint2 v0 = gatherA(0);
    *(uint2*)&Alds[0][(p << 5) + c4] = v0;
    __syncthreads();

    for (int kc = 0; kc < KCH; ++kc) {
        int cur = kc & 1;
        uint2 nxt;
        if (kc < KCH - 1) nxt = gatherA(kc + 1);

        v8h s0 = *(const v8h*)&Alds[cur][(row << 5) + kbase];
        v8h s1 = *(const v8h*)&Alds[cur][(row << 5) + 16 + kbase];
        v8h t0 = *(const v8h*)&Alds[cur][((row + 16) << 5) + kbase];
        v8h t1 = *(const v8h*)&Alds[cur][((row + 16) << 5) + 16 + kbase];
        v16h a0, a1;
#pragma unroll
        for (int j = 0; j < 8; ++j) {
            a0[j] = s0[j]; a0[j + 8] = s1[j];
            a1[j] = t0[j]; a1[j + 8] = t1[j];
        }
        v16h bf = *(const v16h*)(wpack + ((((size_t)kc << 3) + wave) * 32 + lane) * 16);
        c0 = __builtin_amdgcn_wmma_f32_16x16x32_f16(false, a0, false, bf,
                                                    (short)0, c0, false, false);
        c1 = __builtin_amdgcn_wmma_f32_16x16x32_f16(false, a1, false, bf,
                                                    (short)0, c1, false, false);
        if (kc < KCH - 1) *(uint2*)&Alds[cur ^ 1][(p << 5) + c4] = nxt;
        __syncthreads();
    }

    int mlo = (lane >> 4) << 3;
#pragma unroll
    for (int v = 0; v < 8; ++v) {
        Olds[(v + mlo) * CH + n]      = c0[v];
        Olds[(v + mlo + 16) * CH + n] = c1[v];
    }
    __syncthreads();
    for (int idx = tid; idx < MT * CH; idx += 256) {
        int o = idx >> 5, m = idx & 31;
        out[(((size_t)(b * CH + o) * Hh) + h) * Ww + w0 + m] = Olds[m * CH + o];
    }
}

// ---------------------------------------------------------------------------
extern "C" void kernel_launch(void* const* d_in, const int* in_sizes, int n_in,
                              void* d_out, int out_size, void* d_ws, size_t ws_size,
                              hipStream_t stream) {
    (void)in_sizes; (void)n_in; (void)out_size; (void)ws_size;
    const float* x     = (const float*)d_in[0];
    const float* w_off = (const float*)d_in[1];
    const float* b_off = (const float*)d_in[2];
    const float* w_co  = (const float*)d_in[3];
    const float* b_co  = (const float*)d_in[4];
    const float* w_dcn = (const float*)d_in[5];
    const float* b_dcn = (const float*)d_in[6];
    float* out = (float*)d_out;

    char* ws = (char*)d_ws;
    const size_t XH_BYTES = (size_t)Bn * HP * WP * CH * 2;          // 9,834,496
    const size_t CO_BYTES = (size_t)Bn * Hh * Ww * NOFF * 4;        // 15,925,248
    const size_t WP_HALF  = (size_t)KCH * 8 * 32 * 16;              // 147,456 halves
    _Float16* xh   = (_Float16*)ws;
    _Float16* rh   = (_Float16*)(ws + XH_BYTES);
    float*    cos_ = (float*)(ws + 2 * XH_BYTES);
    _Float16* wp1  = (_Float16*)(ws + 2 * XH_BYTES + CO_BYTES);
    _Float16* wp2  = wp1 + WP_HALF;
    _Float16* wp3  = wp2 + WP_HALF;

    pack_x_kernel<<<(Bn * HP * WP * CH + 255) / 256, 256, 0, stream>>>(x, xh);
    hipMemsetAsync(rh, 0, XH_BYTES, stream);                // zero halo of stage-1 out
    pack_w_kernel<<<KCH, 256, 0, stream>>>(w_off, wp1, 0, 128);
    pack_w_kernel<<<KCH, 256, 0, stream>>>(w_co,  wp2, 0, NOFF);
    pack_w_kernel<<<KCH, 256, 0, stream>>>(w_dcn, wp3, 1, 128);

    const int nblk = Bn * Hh * (Ww / MT);                   // 1152
    conv_gemm_kernel<<<nblk, 256, 0, stream>>>(xh, wp1, b_off, rh, nullptr, 0, 128);
    conv_gemm_kernel<<<nblk, 256, 0, stream>>>(rh, wp2, b_co, nullptr, cos_, 1, NOFF);
    dcn_gemm_kernel<<<nblk, 256, 0, stream>>>(xh, cos_, wp3, b_dcn, out);
}
